// FlowGMM_16234976378904
// MI455X (gfx1250) — compile-verified, matrix operand order checked by reading, not run on hardware
//
#include <hip/hip_runtime.h>
#include <hip/hip_bf16.h>
#include <math.h>

typedef unsigned short u16;
typedef __attribute__((ext_vector_type(16))) __bf16 v16bf;
typedef __attribute__((ext_vector_type(8)))  __bf16 v8bf;
typedef __attribute__((ext_vector_type(8)))  float  v8f;

#define NSAMP 65536
#define DFEAT 64
#define DHALF 32
#define HID   512
#define NLAYER 8
#define NCOMP 10
#define LDA 40   // LDS row stride in shorts (32 data + 8 pad)

__device__ __forceinline__ u16 f2bf(float f) {
  union { float f; unsigned u; } v; v.f = f;
  unsigned r = v.u + 0x7fffu + ((v.u >> 16) & 1u);
  return (u16)(r >> 16);
}

// ---------------- weight conversion ----------------
__global__ void cvt_f2bf_kernel(const float* __restrict__ in, u16* __restrict__ out, int n) {
  int i = blockIdx.x * 256 + threadIdx.x;
  if (i < n) out[i] = f2bf(in[i]);
}

// ---------------- init / zero ----------------
__global__ void init_copy_kernel(const float* __restrict__ xin, float* __restrict__ X) {
  int i = blockIdx.x * 256 + threadIdx.x;     // over NSAMP*DFEAT
  int n = i >> 6, f = i & 63;
  X[i] = xin[(size_t)n * (DFEAT + 1) + f];
}

__global__ void zero_kernel(float* __restrict__ p) {
  p[blockIdx.x * 256 + threadIdx.x] = 0.0f;
}

// ---------------- batchnorm reductions ----------------
__global__ void bn_stage1(const float* __restrict__ X, float* __restrict__ part) {
  __shared__ float ss[256], sq[256];
  int t = threadIdx.x;
  int f = t & 63;
  int rg = t >> 6;                              // 0..3
  size_t rowBase = (size_t)blockIdx.x * 256;
  float s0 = 0.f, q0 = 0.f;
  for (int r = rg; r < 256; r += 4) {
    float v = X[(rowBase + r) * DFEAT + f];
    s0 += v; q0 += v * v;
  }
  ss[t] = s0; sq[t] = q0;
  __syncthreads();
  if (t < 64) {
    float a = ss[t] + ss[t + 64] + ss[t + 128] + ss[t + 192];
    float b = sq[t] + sq[t + 64] + sq[t + 128] + sq[t + 192];
    part[blockIdx.x * 128 + f * 2]     = a;
    part[blockIdx.x * 128 + f * 2 + 1] = b;
  }
}

__global__ void bn_stage2(const float* __restrict__ part, float* __restrict__ mu,
                          float* __restrict__ rsig) {
  int f = threadIdx.x;
  if (f < 64) {
    float a = 0.f, b = 0.f;
    for (int blk = 0; blk < 256; ++blk) {
      a += part[blk * 128 + f * 2];
      b += part[blk * 128 + f * 2 + 1];
    }
    float m   = a / 65536.0f;
    float var = b / 65536.0f - m * m;
    mu[f]   = m;
    rsig[f] = 1.0f / sqrtf(var + 1e-5f);
  }
}

__global__ void bn_norm_pack(float* __restrict__ X, const float* __restrict__ mu,
                             const float* __restrict__ rsig, u16* __restrict__ x1pack,
                             int parity) {
  int i = blockIdx.x * 256 + threadIdx.x;      // over NSAMP*DFEAT
  int n = i >> 6, f = i & 63;
  float v = (X[i] - mu[f]) * rsig[f];
  X[i] = v;
  int x1base = parity ? DHALF : 0;
  int j = f - x1base;
  if (j >= 0 && j < DHALF) x1pack[(size_t)n * DHALF + j] = f2bf(v);
}

// ---------------- WMMA GEMM:  C[N,O] = act(A[N,K] * W[O,K]^T + bias) ----------------
// Block: 256 threads = 8 waves. Block tile 256(M) x 64(O).
// Each wave: 2 M-tiles x 4 O-tiles = 8 WMMAs per 32-wide K-step.
// Double-buffered LDS staging: one barrier per K-step; VMEM overlaps WMMA.
__device__ __forceinline__ v16bf load_frag(const u16* __restrict__ base, int lane) {
  // 16-bit A/B fragment layout (ISA 7.12.2):
  // lanes 0-15 : row=lane,    K[0..7]  in v0..v3, K[16..23] in v4..v7
  // lanes 16-31: row=lane-16, K[8..15] in v0..v3, K[24..31] in v4..v7
  int row = lane & 15;
  int k0  = (lane & 16) ? 8 : 0;
  const u16* p = base + row * LDA + k0;
  v8bf lo = *reinterpret_cast<const v8bf*>(p);
  v8bf hi = *reinterpret_cast<const v8bf*>(p + 16);
  v16bf f;
#pragma unroll
  for (int i = 0; i < 8; ++i) { f[i] = lo[i]; f[i + 8] = hi[i]; }
  return f;
}

__device__ __forceinline__ void stage_tiles(
    const u16* __restrict__ A, int K, const u16* __restrict__ W, int O,
    int rowBase, int colBase, int k,
    u16* __restrict__ sA, u16* __restrict__ sW, int tid) {
  // A tile: 256x32 bf16 = 1024 x 16B chunks, 4 per thread
#pragma unroll
  for (int it = 0; it < 4; ++it) {
    int idx = (tid + it * 256) * 8;
    int r = idx >> 5, c = idx & 31;
    v8bf val = *reinterpret_cast<const v8bf*>(A + (size_t)(rowBase + r) * K + k + c);
    *reinterpret_cast<v8bf*>(&sA[r * LDA + c]) = val;
  }
  // W tile: 64x32 bf16, 1 chunk per thread (guard O)
  {
    int idx = tid * 8;
    int r = idx >> 5, c = idx & 31;
    v8bf val = {};
    if (colBase + r < O)
      val = *reinterpret_cast<const v8bf*>(W + (size_t)(colBase + r) * K + k + c);
    *reinterpret_cast<v8bf*>(&sW[r * LDA + c]) = val;
  }
}

__global__ __launch_bounds__(256) void gemm_bias_act(
    const u16* __restrict__ A, int K,
    const u16* __restrict__ W, int O,
    const float* __restrict__ bias,
    u16* __restrict__ outb, float* __restrict__ outf, int relu) {
  __shared__ __align__(16) u16 sA[2][256 * LDA];
  __shared__ __align__(16) u16 sW[2][64 * LDA];

  int rowBase = blockIdx.x * 256;
  int colBase = blockIdx.y * 64;
  int tid  = threadIdx.x;
  int wave = tid >> 5;
  int lane = tid & 31;

  v8f acc[2][4] = {};

  const int nk = K >> 5;
  stage_tiles(A, K, W, O, rowBase, colBase, 0, sA[0], sW[0], tid);

  for (int ki = 0; ki < nk; ++ki) {
    __syncthreads();                           // buf[ki&1] ready; prev compute done
    if (ki + 1 < nk)
      stage_tiles(A, K, W, O, rowBase, colBase, (ki + 1) << 5,
                  sA[(ki + 1) & 1], sW[(ki + 1) & 1], tid);
    if (ki + 2 < nk) {
      int idx = tid * 8;
      int r = idx >> 5, c = idx & 31;
      __builtin_prefetch(A + (size_t)(rowBase + r) * K + ((ki + 2) << 5) + c, 0, 1);
    }
    const u16* cA = sA[ki & 1];
    const u16* cW = sW[ki & 1];
    v16bf a0 = load_frag(&cA[(wave * 32) * LDA], lane);
    v16bf a1 = load_frag(&cA[(wave * 32 + 16) * LDA], lane);
#pragma unroll
    for (int cb = 0; cb < 4; ++cb) {
      v16bf wf = load_frag(&cW[(cb * 16) * LDA], lane);
      acc[0][cb] = __builtin_amdgcn_wmma_f32_16x16x32_bf16(
          false, a0, false, wf, (short)0, acc[0][cb], false, false);
      acc[1][cb] = __builtin_amdgcn_wmma_f32_16x16x32_bf16(
          false, a1, false, wf, (short)0, acc[1][cb], false, false);
    }
  }

  // epilogue: C/D layout -> m = mt*16 + r + (lane>=16 ? 8 : 0), n = cb*16 + (lane&15)
  int nCol = lane & 15;
#pragma unroll
  for (int mt = 0; mt < 2; ++mt) {
    int mBase = rowBase + wave * 32 + mt * 16 + ((lane & 16) ? 8 : 0);
#pragma unroll
    for (int cb = 0; cb < 4; ++cb) {
      int col = colBase + cb * 16 + nCol;
      if (col < O) {
        float bv = bias[col];
#pragma unroll
        for (int r = 0; r < 8; ++r) {
          float v = acc[mt][cb][r] + bv;
          if (relu) v = fmaxf(v, 0.0f);
          size_t oi = (size_t)(mBase + r) * O + col;
          if (outb) outb[oi] = f2bf(v);
          if (outf) outf[oi] = v;
        }
      }
    }
  }
}

// ---------------- coupling update ----------------
__global__ void coupling_kernel(const float* __restrict__ Xn, const float* __restrict__ s,
                                const float* __restrict__ t, float* __restrict__ Xout,
                                float* __restrict__ det, int parity) {
  int n = blockIdx.x * 256 + threadIdx.x;
  const float* xr = Xn + (size_t)n * DFEAT;
  float* xo = Xout + (size_t)n * DFEAT;
  int x1base = parity ? DHALF : 0;
  int x2base = parity ? 0 : DHALF;
  float dsum = 0.f;
#pragma unroll
  for (int j = 0; j < DHALF; ++j) xo[j] = xr[x1base + j];
#pragma unroll
  for (int j = 0; j < DHALF; ++j) {
    float sv = s[(size_t)n * DHALF + j];
    float tv = t[(size_t)n * DHALF + j];
    xo[DHALF + j] = xr[x2base + j] * expf(sv) + tv;
    dsum += sv;
  }
  det[n] += dsum;
}

// ---------------- GMM log-likelihood + copy y ----------------
__global__ void gmm_kernel(const float* __restrict__ Y, const float* __restrict__ means,
                           float* __restrict__ ll, float* __restrict__ yout) {
  __shared__ float sm[NCOMP * DFEAT];
  for (int i = threadIdx.x; i < NCOMP * DFEAT; i += 256) sm[i] = means[i];
  __syncthreads();
  int n = blockIdx.x * 256 + threadIdx.x;
  const float* y = Y + (size_t)n * DFEAT;
  float yr[DFEAT];
#pragma unroll
  for (int f = 0; f < DFEAT; f += 4) {
    float4 v = *reinterpret_cast<const float4*>(y + f);
    yr[f] = v.x; yr[f + 1] = v.y; yr[f + 2] = v.z; yr[f + 3] = v.w;
    *reinterpret_cast<float4*>(yout + (size_t)n * DFEAT + f) = v;
  }
  float lp[NCOMP];
  float mx = -3.4e38f;
  for (int k = 0; k < NCOMP; ++k) {
    float ssum = 0.f;
    for (int f = 0; f < DFEAT; ++f) {
      float d = yr[f] - sm[k * DFEAT + f];
      ssum += d * d;
    }
    float v = -0.5f * (ssum + 64.0f * 1.8378770664093453f) - 2.302585092994046f; // - ln K
    lp[k] = v;
    mx = fmaxf(mx, v);
  }
  float se = 0.f;
  for (int k = 0; k < NCOMP; ++k) se += expf(lp[k] - mx);
  ll[n] = mx + logf(se);
}

// ---------------- final reductions ----------------
__global__ void red1_kernel(const float* __restrict__ ll, const float* __restrict__ det,
                            float* __restrict__ part) {
  __shared__ float a[256], b[256];
  int t = threadIdx.x;
  int i = blockIdx.x * 256 + t;
  a[t] = ll[i]; b[t] = det[i];
  __syncthreads();
  for (int s = 128; s > 0; s >>= 1) {
    if (t < s) { a[t] += a[t + s]; b[t] += b[t + s]; }
    __syncthreads();
  }
  if (t == 0) { part[blockIdx.x * 2] = a[0]; part[blockIdx.x * 2 + 1] = b[0]; }
}

__global__ void red2_kernel(const float* __restrict__ part, float* __restrict__ loss,
                            float* __restrict__ llm, float* __restrict__ dsm) {
  __shared__ float a[256], b[256];
  int t = threadIdx.x;
  a[t] = part[t * 2]; b[t] = part[t * 2 + 1];
  __syncthreads();
  for (int s = 128; s > 0; s >>= 1) {
    if (t < s) { a[t] += a[t + s]; b[t] += b[t + s]; }
    __syncthreads();
  }
  if (t == 0) {
    float lm = a[0] / 65536.0f;
    float dm = b[0] / 65536.0f;
    *llm = lm; *dsm = dm; *loss = -(lm + dm);
  }
}

// ---------------- host orchestration ----------------
extern "C" void kernel_launch(void* const* d_in, const int* in_sizes, int n_in,
                              void* d_out, int out_size, void* d_ws, size_t ws_size,
                              hipStream_t stream) {
  const float* xin  = (const float*)d_in[0];
  const float* sW0f = (const float*)d_in[1];
  const float* sb0  = (const float*)d_in[2];
  const float* sW1f = (const float*)d_in[3];
  const float* sb1  = (const float*)d_in[4];
  const float* sW2f = (const float*)d_in[5];
  const float* sb2  = (const float*)d_in[6];
  const float* tW0f = (const float*)d_in[7];
  const float* tb0  = (const float*)d_in[8];
  const float* tW1f = (const float*)d_in[9];
  const float* tb1  = (const float*)d_in[10];
  const float* tW2f = (const float*)d_in[11];
  const float* tb2  = (const float*)d_in[12];
  const float* means = (const float*)d_in[13];

  // output layout: y[N*D], loss[1], ll[N], det[N], ll_mean[1], det_mean[1]
  float* out      = (float*)d_out;
  float* y_out    = out;
  float* loss_out = out + (size_t)NSAMP * DFEAT;
  float* ll_out   = loss_out + 1;
  float* det_out  = ll_out + NSAMP;
  float* llm_out  = det_out + NSAMP;
  float* dsm_out  = llm_out + 1;

  // workspace carve-up (256B aligned)
  size_t off = 0;
  char* base = (char*)d_ws;
  auto alloc = [&](size_t bytes) -> void* {
    void* p = base + off;
    off = (off + bytes + 255) & ~(size_t)255;
    return p;
  };
  float* X0     = (float*)alloc((size_t)NSAMP * DFEAT * 4);
  float* X1     = (float*)alloc((size_t)NSAMP * DFEAT * 4);
  u16*   x1pack = (u16*)  alloc((size_t)NSAMP * DHALF * 2);
  u16*   h0     = (u16*)  alloc((size_t)NSAMP * HID * 2);
  u16*   h1     = (u16*)  alloc((size_t)NSAMP * HID * 2);
  float* sbuf   = (float*)alloc((size_t)NSAMP * DHALF * 4);
  float* tbuf   = (float*)alloc((size_t)NSAMP * DHALF * 4);
  const int W0N = NLAYER * HID * DHALF;    // 131072
  const int W1N = NLAYER * HID * HID;      // 2097152
  const int W2N = NLAYER * DHALF * HID;    // 131072
  u16* sW0b = (u16*)alloc((size_t)W0N * 2);
  u16* sW1b = (u16*)alloc((size_t)W1N * 2);
  u16* sW2b = (u16*)alloc((size_t)W2N * 2);
  u16* tW0b = (u16*)alloc((size_t)W0N * 2);
  u16* tW1b = (u16*)alloc((size_t)W1N * 2);
  u16* tW2b = (u16*)alloc((size_t)W2N * 2);
  float* bnpart = (float*)alloc(256 * 128 * 4);
  float* mu     = (float*)alloc(64 * 4);
  float* rsig   = (float*)alloc(64 * 4);
  float* rpart  = (float*)alloc(256 * 2 * 4);

  // convert weights to bf16
  cvt_f2bf_kernel<<<(W0N + 255) / 256, 256, 0, stream>>>(sW0f, sW0b, W0N);
  cvt_f2bf_kernel<<<(W1N + 255) / 256, 256, 0, stream>>>(sW1f, sW1b, W1N);
  cvt_f2bf_kernel<<<(W2N + 255) / 256, 256, 0, stream>>>(sW2f, sW2b, W2N);
  cvt_f2bf_kernel<<<(W0N + 255) / 256, 256, 0, stream>>>(tW0f, tW0b, W0N);
  cvt_f2bf_kernel<<<(W1N + 255) / 256, 256, 0, stream>>>(tW1f, tW1b, W1N);
  cvt_f2bf_kernel<<<(W2N + 255) / 256, 256, 0, stream>>>(tW2f, tW2b, W2N);

  init_copy_kernel<<<NSAMP * DFEAT / 256, 256, 0, stream>>>(xin, X0);
  zero_kernel<<<NSAMP / 256, 256, 0, stream>>>(det_out);

  float* Xcur = X0;
  float* Xnxt = X1;
  const dim3 gFull(NSAMP / 256, HID / 64);   // (256, 8)
  const dim3 gLast(NSAMP / 256, 1);          // (256, 1)

  for (int i = 0; i < NLAYER; ++i) {
    int parity = i & 1;
    bn_stage1<<<256, 256, 0, stream>>>(Xcur, bnpart);
    bn_stage2<<<1, 64, 0, stream>>>(bnpart, mu, rsig);
    bn_norm_pack<<<NSAMP * DFEAT / 256, 256, 0, stream>>>(Xcur, mu, rsig, x1pack, parity);

    // s-MLP
    gemm_bias_act<<<gFull, 256, 0, stream>>>(x1pack, DHALF, sW0b + (size_t)i * HID * DHALF, HID,
                                             sb0 + (size_t)i * HID, h0, nullptr, 1);
    gemm_bias_act<<<gFull, 256, 0, stream>>>(h0, HID, sW1b + (size_t)i * HID * HID, HID,
                                             sb1 + (size_t)i * HID, h1, nullptr, 1);
    gemm_bias_act<<<gLast, 256, 0, stream>>>(h1, HID, sW2b + (size_t)i * DHALF * HID, DHALF,
                                             sb2 + (size_t)i * DHALF, nullptr, sbuf, 0);
    // t-MLP
    gemm_bias_act<<<gFull, 256, 0, stream>>>(x1pack, DHALF, tW0b + (size_t)i * HID * DHALF, HID,
                                             tb0 + (size_t)i * HID, h0, nullptr, 1);
    gemm_bias_act<<<gFull, 256, 0, stream>>>(h0, HID, tW1b + (size_t)i * HID * HID, HID,
                                             tb1 + (size_t)i * HID, h1, nullptr, 1);
    gemm_bias_act<<<gLast, 256, 0, stream>>>(h1, HID, tW2b + (size_t)i * DHALF * HID, DHALF,
                                             tb2 + (size_t)i * DHALF, nullptr, tbuf, 0);

    coupling_kernel<<<NSAMP / 256, 256, 0, stream>>>(Xcur, sbuf, tbuf, Xnxt, det_out, parity);
    float* tmp = Xcur; Xcur = Xnxt; Xnxt = tmp;
  }

  gmm_kernel<<<NSAMP / 256, 256, 0, stream>>>(Xcur, means, ll_out, y_out);
  red1_kernel<<<256, 256, 0, stream>>>(ll_out, det_out, rpart);
  red2_kernel<<<1, 256, 0, stream>>>(rpart, loss_out, llm_out, dsm_out);
}
